// RegionDetectionRCNN_50079318672093
// MI455X (gfx1250) — matrix operand order repaired and checked by reading, not compiled, not run on hardware
//
#include <hip/hip_runtime.h>
#include <hip/hip_bf16.h>

typedef __attribute__((ext_vector_type(8)))  float        v8f;
typedef __attribute__((ext_vector_type(8)))  __bf16       v8bf;
typedef __attribute__((ext_vector_type(16))) __bf16       v16bf;
typedef __attribute__((ext_vector_type(4)))  unsigned int v4u;
typedef __attribute__((ext_vector_type(4)))  int          v4i;
typedef __bf16 bf16;

#define FH_ 100
#define FW_ 100
#define CIN 256
#define NB  2
#define NPIX (NB*FH_*FW_)          // 20000
#define KCONV (9*CIN)              // 2304
#define NANCH 30000                // 3*100*100
#define PRE_NMS_ 1000
#define POST_NMS_ 512
#define NROI (NB*POST_NMS_)        // 1024
#define KFC1 12544                 // 256*7*7
#define NHID 1024
#define DWH_CLAMP_ 4.135166556742356f

// ---- gfx1250 async global->LDS copy (ASYNCcnt path), with safe fallback ----
#if defined(__has_builtin)
#if __has_builtin(__builtin_amdgcn_global_load_async_to_lds_b128)
#define ASYNC_LDS 1
#endif
#endif
#ifndef ASYNC_LDS
#define ASYNC_LDS 0
#endif

__device__ __forceinline__ void copy16_g2l(bf16* dst, const bf16* src) {
#if ASYNC_LDS
    // flat->LDS: LDS address is the low 32 bits of the flat address (ISA 10.2); flat->global value-identical.
    auto g = (__attribute__((address_space(1))) v4i*)(unsigned long long)src;
    auto l = (__attribute__((address_space(3))) v4i*)(unsigned int)(unsigned long long)dst;
    __builtin_amdgcn_global_load_async_to_lds_b128(g, l, 0, 0);
#else
    *(v4u*)dst = *(const v4u*)src;
#endif
}

__device__ __forceinline__ void tile_barrier_after_fill() {
#if ASYNC_LDS
    asm volatile("s_wait_asynccnt 0x0" ::: "memory");
#endif
    __syncthreads();
}

__device__ __forceinline__ v16bf make_v16(const bf16* lo, const bf16* hi) {
    v8bf a = *(const v8bf*)lo;
    v8bf b = *(const v8bf*)hi;
    v16bf r;
#pragma unroll
    for (int i = 0; i < 8; ++i) { r[i] = a[i]; r[i + 8] = b[i]; }
    return r;
}

// ---------------- pack kernels ----------------
__global__ void pack_feat_nhwc(const float* __restrict__ src, bf16* __restrict__ dst) {
    int i = blockIdx.x * blockDim.x + threadIdx.x;       // over B*C*H*W
    if (i >= NB * CIN * FH_ * FW_) return;
    int b = i / (CIN * FH_ * FW_);
    int rem = i % (CIN * FH_ * FW_);
    int c = rem / (FH_ * FW_);
    int p = rem % (FH_ * FW_);
    dst[((size_t)(b * FH_ * FW_ + p)) * CIN + c] = (bf16)src[i];
}

__global__ void pack_rpn_w(const float* __restrict__ src, bf16* __restrict__ dst) {
    int i = blockIdx.x * blockDim.x + threadIdx.x;       // over O*I*9 (OIHW)
    if (i >= CIN * KCONV) return;
    int o = i / KCONV;
    int rem = i % KCONV;
    int c = rem / 9;
    int tap = rem % 9;
    dst[(size_t)o * KCONV + tap * CIN + c] = (bf16)src[i];
}

__global__ void pack_transpose_bf16(const float* __restrict__ src, bf16* __restrict__ dst,
                                    int K, int N) {   // src [K][N] -> dst [N][K]
    int i = blockIdx.x * blockDim.x + threadIdx.x;
    if (i >= K * N) return;
    int k = i / N, n = i % N;
    dst[(size_t)n * K + k] = (bf16)src[i];
}

// ============ WMMA micro-tile: 16 waves, each wave owns 32x32 (4 WMMAs/K-step) ============
// Block tile 128x128, BK=32. As/Bs: [128][32] bf16.

__device__ __forceinline__ void wave_mma_step(const bf16* As, const bf16* Bs,
                                              int wm, int wn, int lane,
                                              v8f& a00, v8f& a01, v8f& a10, v8f& a11) {
    const int rl = lane & 15;
    const int kh = lane >> 4;        // which K-half this lane holds
    v16bf af0 = make_v16(&As[(wm * 32 + rl) * 32 + kh * 8],
                         &As[(wm * 32 + rl) * 32 + 16 + kh * 8]);
    v16bf af1 = make_v16(&As[(wm * 32 + 16 + rl) * 32 + kh * 8],
                         &As[(wm * 32 + 16 + rl) * 32 + 16 + kh * 8]);
    v16bf bf0 = make_v16(&Bs[(wn * 32 + rl) * 32 + kh * 8],
                         &Bs[(wn * 32 + rl) * 32 + 16 + kh * 8]);
    v16bf bf1 = make_v16(&Bs[(wn * 32 + 16 + rl) * 32 + kh * 8],
                         &Bs[(wn * 32 + 16 + rl) * 32 + 16 + kh * 8]);
    a00 = __builtin_amdgcn_wmma_f32_16x16x32_bf16(false, af0, false, bf0, (short)0, a00, false, false);
    a01 = __builtin_amdgcn_wmma_f32_16x16x32_bf16(false, af0, false, bf1, (short)0, a01, false, false);
    a10 = __builtin_amdgcn_wmma_f32_16x16x32_bf16(false, af1, false, bf0, (short)0, a10, false, false);
    a11 = __builtin_amdgcn_wmma_f32_16x16x32_bf16(false, af1, false, bf1, (short)0, a11, false, false);
}

// ---------------- implicit-GEMM 3x3 conv (WMMA bf16) ----------------
// F: [B][H][W][Cin] bf16, Wp: [O][tap*256+c] bf16, T out: [B][H][W][O] bf16 (relu(conv+bias))
__global__ __launch_bounds__(512)
void conv3x3_wmma(const bf16* __restrict__ F, const bf16* __restrict__ Wp,
                  const float* __restrict__ bias, bf16* __restrict__ T) {
    __shared__ __align__(32) bf16 As[128 * 32];
    __shared__ __align__(32) bf16 Bs[128 * 32];
    const int tid = threadIdx.x;
    const int lane = tid & 31;
    const int wave = tid >> 5;            // 0..15
    const int wm = wave >> 2, wn = wave & 3;
    const int mBase = blockIdx.x * 128;   // pixel base
    const int nBase = blockIdx.y * 128;   // out-channel base
    v8f a00 = {}, a01 = {}, a10 = {}, a11 = {};

    const int ldrow = tid >> 2;           // 0..127
    const int ldseg = (tid & 3) * 8;      // 0,8,16,24 (8 bf16 = 16B per thread)

    // per-row pixel geometry (invariant over K)
    const int pm = mBase + ldrow;
    const bool rowOK = pm < NPIX;
    const int pb = pm / (FH_ * FW_);
    const int pp = pm % (FH_ * FW_);
    const int py = pp / FW_, px = pp % FW_;
    const bf16* wRow = Wp + (size_t)(nBase + ldrow) * KCONV + ldseg;  // o < 256 always

    for (int k0 = 0; k0 < KCONV; k0 += 32) {
        const int tap = k0 >> 8;          // 32-chunk always inside one tap (256 % 32 == 0)
        const int cbase = (k0 & 255) + ldseg;
        const int iy = py + tap / 3 - 1;
        const int ix = px + tap % 3 - 1;
        bf16* aDst = &As[ldrow * 32 + ldseg];
        if (rowOK && iy >= 0 && iy < FH_ && ix >= 0 && ix < FW_) {
            copy16_g2l(aDst, F + (((size_t)pb * FH_ + iy) * FW_ + ix) * CIN + cbase);
        } else {
            v4u z = {0u, 0u, 0u, 0u};
            *(v4u*)aDst = z;              // zero-pad via plain LDS store
        }
        copy16_g2l(&Bs[ldrow * 32 + ldseg], wRow + k0);
        tile_barrier_after_fill();
        wave_mma_step(As, Bs, wm, wn, lane, a00, a01, a10, a11);
        __syncthreads();
    }

    // epilogue: bias + relu -> bf16 NHWC
    const v8f accs[4] = {a00, a01, a10, a11};
#pragma unroll
    for (int sm = 0; sm < 2; ++sm)
#pragma unroll
        for (int sn = 0; sn < 2; ++sn) {
            v8f acc = accs[sm * 2 + sn];
            int o = nBase + wn * 32 + sn * 16 + (lane & 15);
            float bv = bias[o];
#pragma unroll
            for (int r8 = 0; r8 < 8; ++r8) {
                int m = mBase + wm * 32 + sm * 16 + r8 + 8 * (lane >> 4);
                if (m < NPIX) {
                    float v = acc[r8] + bv;
                    v = v > 0.f ? v : 0.f;
                    T[(size_t)m * CIN + o] = (bf16)v;
                }
            }
        }
}

// ---------------- generic GEMM (WMMA bf16): C = relu(A*Bt^T + bias) ----------------
// A [M][K] bf16, Bt [N][K] bf16, out [M][N] bf16. K%32==0, M,N % 128 == 0.
__global__ __launch_bounds__(512)
void gemm_bf16_wmma(const bf16* __restrict__ A, const bf16* __restrict__ Bt,
                    const float* __restrict__ bias, bf16* __restrict__ C,
                    int M, int N, int K, int relu) {
    __shared__ __align__(32) bf16 As[128 * 32];
    __shared__ __align__(32) bf16 Bs[128 * 32];
    const int tid = threadIdx.x;
    const int lane = tid & 31;
    const int wave = tid >> 5;
    const int wm = wave >> 2, wn = wave & 3;
    const int mBase = blockIdx.x * 128;
    const int nBase = blockIdx.y * 128;
    v8f a00 = {}, a01 = {}, a10 = {}, a11 = {};

    const int ldrow = tid >> 2;           // 0..127
    const int ldseg = (tid & 3) * 8;      // 16B per thread
    const bf16* aRow = A + (size_t)(mBase + ldrow) * K + ldseg;
    const bf16* bRow = Bt + (size_t)(nBase + ldrow) * K + ldseg;

    for (int k0 = 0; k0 < K; k0 += 32) {
        copy16_g2l(&As[ldrow * 32 + ldseg], aRow + k0);
        copy16_g2l(&Bs[ldrow * 32 + ldseg], bRow + k0);
        tile_barrier_after_fill();
        wave_mma_step(As, Bs, wm, wn, lane, a00, a01, a10, a11);
        __syncthreads();
    }

    const v8f accs[4] = {a00, a01, a10, a11};
#pragma unroll
    for (int sm = 0; sm < 2; ++sm)
#pragma unroll
        for (int sn = 0; sn < 2; ++sn) {
            v8f acc = accs[sm * 2 + sn];
            int n = nBase + wn * 32 + sn * 16 + (lane & 15);
            float bv = bias ? bias[n] : 0.f;
#pragma unroll
            for (int r8 = 0; r8 < 8; ++r8) {
                int m = mBase + wm * 32 + sm * 16 + r8 + 8 * (lane >> 4);
                float v = acc[r8] + bv;
                if (relu) v = v > 0.f ? v : 0.f;
                C[(size_t)m * N + n] = (bf16)v;
            }
        }
}

// ---------------- RPN 1x1 heads (cls 3 + box 12) ----------------
__global__ __launch_bounds__(128)
void rpn_heads(const bf16* __restrict__ T, const float* __restrict__ cw,
               const float* __restrict__ cb, const float* __restrict__ bw,
               const float* __restrict__ bb, float* __restrict__ scores,
               float* __restrict__ deltas) {
    __shared__ float Wl[15 * 256];
    int tid = threadIdx.x;
    for (int i = tid; i < 15 * 256; i += blockDim.x) {
        int o = i / 256, c = i % 256;
        Wl[i] = (o < 3) ? cw[o * 256 + c] : bw[(o - 3) * 256 + c];
    }
    __syncthreads();
    int pm = blockIdx.x * blockDim.x + tid;
    if (pm >= NPIX) return;
    int b = pm / (FH_ * FW_);
    int p = pm % (FH_ * FW_);
    const bf16* tp = T + (size_t)pm * CIN;
    float acc[15];
#pragma unroll
    for (int o = 0; o < 15; ++o) acc[o] = 0.f;
    for (int c = 0; c < 256; ++c) {
        float tv = (float)tp[c];
#pragma unroll
        for (int o = 0; o < 15; ++o) acc[o] += tv * Wl[o * 256 + c];
    }
#pragma unroll
    for (int a = 0; a < 3; ++a) scores[b * NANCH + p * 3 + a] = acc[a] + cb[a];
#pragma unroll
    for (int o = 0; o < 12; ++o)
        deltas[((size_t)b * NANCH + p * 3 + (o >> 2)) * 4 + (o & 3)] = acc[3 + o] + bb[o];
}

// ---------------- anchor decode ----------------
__global__ void decode_boxes(const float* __restrict__ deltas, const float* __restrict__ anchors,
                             float* __restrict__ boxes) {
    int i = blockIdx.x * blockDim.x + threadIdx.x;   // over B*NANCH
    if (i >= NB * NANCH) return;
    int a = i % NANCH;
    const float* d = deltas + (size_t)i * 4;
    float x1 = anchors[a * 4 + 0], y1 = anchors[a * 4 + 1];
    float x2 = anchors[a * 4 + 2], y2 = anchors[a * 4 + 3];
    float wa = x2 - x1, ha = y2 - y1;
    float cxa = x1 + 0.5f * wa, cya = y1 + 0.5f * ha;
    float dx = d[0], dy = d[1];
    float dw = fminf(d[2], DWH_CLAMP_), dh = fminf(d[3], DWH_CLAMP_);
    float cx = dx * wa + cxa, cy = dy * ha + cya;
    float w = wa * __expf(dw), h = ha * __expf(dh);
    float* o = boxes + (size_t)i * 4;
    o[0] = fminf(fmaxf(cx - 0.5f * w, 0.f), 800.f);
    o[1] = fminf(fmaxf(cy - 0.5f * h, 0.f), 800.f);
    o[2] = fminf(fmaxf(cx + 0.5f * w, 0.f), 800.f);
    o[3] = fminf(fmaxf(cy + 0.5f * h, 0.f), 800.f);
}

// ---------------- per-image top-1000 (iterative argmax) ----------------
__global__ __launch_bounds__(1024)
void topk_select(const float* __restrict__ scores, const float* __restrict__ boxes,
                 float* __restrict__ scwork, float* __restrict__ tk_s, int* __restrict__ tk_i) {
    int img = blockIdx.x, tid = threadIdx.x;
    const float* s = scores + (size_t)img * NANCH;
    float* w = scwork + (size_t)img * NANCH;
    for (int i = tid; i < NANCH; i += 1024) w[i] = s[i];
    __syncthreads();
    __shared__ float sv[1024];
    __shared__ int si[1024];
    for (int k = 0; k < PRE_NMS_; ++k) {
        float best = -3e38f; int bi = 0;
        for (int i = tid; i < NANCH; i += 1024) {
            float v = w[i];
            if (v > best) { best = v; bi = i; }
        }
        sv[tid] = best; si[tid] = bi;
        __syncthreads();
        for (int off = 512; off > 0; off >>= 1) {
            if (tid < off) {
                float a = sv[tid], b = sv[tid + off];
                if (b > a || (b == a && si[tid + off] < si[tid])) { sv[tid] = b; si[tid] = si[tid + off]; }
            }
            __syncthreads();
        }
        if (tid == 0) {
            int idx = si[0];
            const float* bx = boxes + ((size_t)img * NANCH + idx) * 4;
            bool valid = (bx[2] - bx[0] >= 0.01f) && (bx[3] - bx[1] >= 0.01f);
            tk_s[img * 1024 + k] = valid ? sv[0] : -1e9f;
            tk_i[img * 1024 + k] = idx;
            w[idx] = -3e38f;
        }
        __syncthreads();
    }
    if (tid >= PRE_NMS_) { tk_s[img * 1024 + tid] = -3e38f; tk_i[img * 1024 + tid] = 0; }
}

// ---------------- bitonic sort (desc) + gather boxes ----------------
__global__ __launch_bounds__(1024)
void sort1024(const float* __restrict__ tk_s, const int* __restrict__ tk_i,
              const float* __restrict__ boxes, float* __restrict__ sb) {
    int img = blockIdx.x, tid = threadIdx.x;
    __shared__ float s[1024];
    __shared__ int id[1024];
    s[tid] = tk_s[img * 1024 + tid];
    id[tid] = tk_i[img * 1024 + tid];
    __syncthreads();
    for (int size = 2; size <= 1024; size <<= 1) {
        for (int stride = size >> 1; stride > 0; stride >>= 1) {
            int j = tid ^ stride;
            if (j > tid) {
                bool ascend = (tid & size) != 0;   // overall descending
                float a = s[tid], b = s[j];
                bool sw = ascend ? (a > b) : (a < b);
                if (sw) {
                    s[tid] = b; s[j] = a;
                    int t = id[tid]; id[tid] = id[j]; id[j] = t;
                }
            }
            __syncthreads();
        }
    }
    const float* bx = boxes + ((size_t)img * NANCH + id[tid]) * 4;
    float* o = sb + ((size_t)img * 1024 + tid) * 4;
    o[0] = bx[0]; o[1] = bx[1]; o[2] = bx[2]; o[3] = bx[3];
}

// ---------------- NMS (sequential suppression) + compact to 512 rois ----------------
__global__ __launch_bounds__(1024)
void nms_compact(const float* __restrict__ sb, float* __restrict__ rois) {
    int img = blockIdx.x, tid = threadIdx.x;
    __shared__ float bx[PRE_NMS_ * 4];
    __shared__ int keep[PRE_NMS_];
    for (int i = tid; i < PRE_NMS_; i += 1024) {
        const float* p = sb + ((size_t)img * 1024 + i) * 4;
        bx[i * 4 + 0] = p[0]; bx[i * 4 + 1] = p[1];
        bx[i * 4 + 2] = p[2]; bx[i * 4 + 3] = p[3];
        keep[i] = 1;
    }
    __syncthreads();
    for (int i = 0; i < PRE_NMS_ - 1; ++i) {
        if (keep[i]) {
            float x1 = bx[i * 4], y1 = bx[i * 4 + 1], x2 = bx[i * 4 + 2], y2 = bx[i * 4 + 3];
            float ai = fmaxf(x2 - x1, 0.f) * fmaxf(y2 - y1, 0.f);
            for (int j = i + 1 + tid; j < PRE_NMS_; j += 1024) {
                if (keep[j]) {
                    float u1 = bx[j * 4], v1 = bx[j * 4 + 1], u2 = bx[j * 4 + 2], v2 = bx[j * 4 + 3];
                    float aj = fmaxf(u2 - u1, 0.f) * fmaxf(v2 - v1, 0.f);
                    float lx = fmaxf(x1, u1), ly = fmaxf(y1, v1);
                    float rx = fminf(x2, u2), ry = fminf(y2, v2);
                    float iw = fmaxf(rx - lx, 0.f), ih = fmaxf(ry - ly, 0.f);
                    float inter = iw * ih;
                    float iou = inter / fmaxf(ai + aj - inter, 1e-6f);
                    if (iou > 0.7f) keep[j] = 0;
                }
            }
        }
        __syncthreads();
    }
    if (tid == 0) {
        int cnt = 0;
        for (int i = 0; i < PRE_NMS_ && cnt < POST_NMS_; ++i)
            if (keep[i]) {
                float* o = rois + ((size_t)img * POST_NMS_ + cnt) * 4;
                o[0] = bx[i * 4]; o[1] = bx[i * 4 + 1]; o[2] = bx[i * 4 + 2]; o[3] = bx[i * 4 + 3];
                ++cnt;
            }
        for (int i = 0; i < PRE_NMS_ && cnt < POST_NMS_; ++i)
            if (!keep[i]) {
                float* o = rois + ((size_t)img * POST_NMS_ + cnt) * 4;
                o[0] = bx[i * 4]; o[1] = bx[i * 4 + 1]; o[2] = bx[i * 4 + 2]; o[3] = bx[i * 4 + 3];
                ++cnt;
            }
    }
}

// ---------------- ROI align (7x7, ratio 2), output [R][c*49+bin] bf16 ----------------
__global__ __launch_bounds__(256)
void roi_align(const bf16* __restrict__ F, const float* __restrict__ rois,
               bf16* __restrict__ pooled) {
    int r = blockIdx.x;          // 0..1023
    int c = threadIdx.x;         // 0..255 (channel)
    int b = r >> 9;              // r / 512
    __shared__ float rx1, ry1, rww, rhh;
    __shared__ float lx[14], ly[14];
    __shared__ int x0s[14], x1s[14], y0s[14], y1s[14];
    if (c == 0) {
        float a0 = rois[r * 4 + 0] * 0.125f, a1 = rois[r * 4 + 1] * 0.125f;
        float a2 = rois[r * 4 + 2] * 0.125f, a3 = rois[r * 4 + 3] * 0.125f;
        rx1 = a0; ry1 = a1;
        rww = fmaxf(a2 - a0, 1.f); rhh = fmaxf(a3 - a1, 1.f);
    }
    __syncthreads();
    if (c < 14) {
        float frac = (c + 0.5f) / 14.f;
        float xv = fminf(fmaxf(rx1 + frac * rww, 0.f), FW_ - 1.f);
        float yv = fminf(fmaxf(ry1 + frac * rhh, 0.f), FH_ - 1.f);
        int xi = (int)floorf(xv), yi = (int)floorf(yv);
        x0s[c] = xi; y0s[c] = yi;
        x1s[c] = min(xi + 1, FW_ - 1); y1s[c] = min(yi + 1, FH_ - 1);
        lx[c] = xv - (float)xi; ly[c] = yv - (float)yi;
    }
    __syncthreads();
    const bf16* Fb = F + (size_t)b * FH_ * FW_ * CIN + c;
    for (int ph = 0; ph < 7; ++ph)
        for (int pw = 0; pw < 7; ++pw) {
            float sum = 0.f;
#pragma unroll
            for (int sy = 0; sy < 2; ++sy)
#pragma unroll
                for (int sx = 0; sx < 2; ++sx) {
                    int iy = ph * 2 + sy, ix = pw * 2 + sx;
                    float w1y = ly[iy], w0y = 1.f - w1y;
                    float w1x = lx[ix], w0x = 1.f - w1x;
                    int yA = y0s[iy], yB = y1s[iy], xA = x0s[ix], xB = x1s[ix];
                    float v00 = (float)Fb[((size_t)yA * FW_ + xA) * CIN];
                    float v01 = (float)Fb[((size_t)yA * FW_ + xB) * CIN];
                    float v10 = (float)Fb[((size_t)yB * FW_ + xA) * CIN];
                    float v11 = (float)Fb[((size_t)yB * FW_ + xB) * CIN];
                    sum += v00 * w0y * w0x + v01 * w0y * w1x + v10 * w1y * w0x + v11 * w1y * w1x;
                }
            pooled[(size_t)r * KFC1 + c * 49 + ph * 7 + pw] = (bf16)(sum * 0.25f);
        }
}

// ---------------- final heads: out[r][0:3]=cls, [3:15]=box ----------------
__global__ __launch_bounds__(256)
void final_heads(const bf16* __restrict__ H, const float* __restrict__ cw,
                 const float* __restrict__ cb, const float* __restrict__ bw,
                 const float* __restrict__ bb, float* __restrict__ out) {
    int r = blockIdx.x * 256 + threadIdx.x;
    if (r >= NROI) return;
    const bf16* h = H + (size_t)r * NHID;
    float acc[15];
#pragma unroll
    for (int o = 0; o < 15; ++o) acc[o] = 0.f;
    for (int k = 0; k < NHID; ++k) {
        float hv = (float)h[k];
#pragma unroll
        for (int o = 0; o < 3; ++o) acc[o] += hv * cw[k * 3 + o];
#pragma unroll
        for (int o = 0; o < 12; ++o) acc[3 + o] += hv * bw[k * 12 + o];
    }
#pragma unroll
    for (int o = 0; o < 3; ++o) out[(size_t)r * 15 + o] = acc[o] + cb[o];
#pragma unroll
    for (int o = 0; o < 12; ++o) out[(size_t)r * 15 + 3 + o] = acc[3 + o] + bb[o];
}

// ---------------- host launch ----------------
extern "C" void kernel_launch(void* const* d_in, const int* in_sizes, int n_in,
                              void* d_out, int out_size, void* d_ws, size_t ws_size,
                              hipStream_t stream) {
    const float* feature    = (const float*)d_in[0];
    const float* anchors    = (const float*)d_in[1];
    const float* rpn_conv_w = (const float*)d_in[2];
    const float* rpn_conv_b = (const float*)d_in[3];
    const float* rpn_cls_w  = (const float*)d_in[4];
    const float* rpn_cls_b  = (const float*)d_in[5];
    const float* rpn_box_w  = (const float*)d_in[6];
    const float* rpn_box_b  = (const float*)d_in[7];
    const float* fc1_w      = (const float*)d_in[8];
    const float* fc1_b      = (const float*)d_in[9];
    const float* fc2_w      = (const float*)d_in[10];
    const float* fc2_b      = (const float*)d_in[11];
    const float* cls_w      = (const float*)d_in[12];
    const float* cls_b      = (const float*)d_in[13];
    const float* box_w      = (const float*)d_in[14];
    const float* box_b      = (const float*)d_in[15];
    float* out = (float*)d_out;

    char* w = (char*)d_ws;
    auto carve = [&](size_t bytes) -> char* {
        char* p = w;
        w += (bytes + 255) & ~(size_t)255;
        return p;
    };
    bf16* feat_nhwc = (bf16*)carve((size_t)NB * FH_ * FW_ * CIN * 2);
    bf16* wpack     = (bf16*)carve((size_t)CIN * KCONV * 2);
    bf16* t_nhwc    = (bf16*)carve((size_t)NPIX * CIN * 2);
    float* scores   = (float*)carve((size_t)NB * NANCH * 4);
    float* deltas   = (float*)carve((size_t)NB * NANCH * 4 * 4);
    float* boxes    = (float*)carve((size_t)NB * NANCH * 4 * 4);
    float* scwork   = (float*)carve((size_t)NB * NANCH * 4);
    float* tk_s     = (float*)carve((size_t)NB * 1024 * 4);
    int*   tk_i     = (int*)carve((size_t)NB * 1024 * 4);
    float* sortedb  = (float*)carve((size_t)NB * 1024 * 4 * 4);
    float* rois     = (float*)carve((size_t)NROI * 4 * 4);
    bf16* pooled    = (bf16*)carve((size_t)NROI * KFC1 * 2);
    bf16* fc1_wt    = (bf16*)carve((size_t)NHID * KFC1 * 2);
    bf16* h1        = (bf16*)carve((size_t)NROI * NHID * 2);
    bf16* fc2_wt    = (bf16*)carve((size_t)NHID * NHID * 2);
    bf16* h2        = (bf16*)carve((size_t)NROI * NHID * 2);

    // 1) pack inputs to bf16 layouts
    pack_feat_nhwc<<<(NB * CIN * FH_ * FW_ + 255) / 256, 256, 0, stream>>>(feature, feat_nhwc);
    pack_rpn_w<<<(CIN * KCONV + 255) / 256, 256, 0, stream>>>(rpn_conv_w, wpack);
    pack_transpose_bf16<<<(KFC1 * NHID + 255) / 256, 256, 0, stream>>>(fc1_w, fc1_wt, KFC1, NHID);
    pack_transpose_bf16<<<(NHID * NHID + 255) / 256, 256, 0, stream>>>(fc2_w, fc2_wt, NHID, NHID);

    // 2) RPN conv (WMMA implicit GEMM, M=20000 N=256 K=2304), 128x128 tiles
    conv3x3_wmma<<<dim3((NPIX + 127) / 128, CIN / 128), 512, 0, stream>>>(feat_nhwc, wpack,
                                                                          rpn_conv_b, t_nhwc);

    // 3) 1x1 heads, decode
    rpn_heads<<<(NPIX + 127) / 128, 128, 0, stream>>>(t_nhwc, rpn_cls_w, rpn_cls_b,
                                                      rpn_box_w, rpn_box_b, scores, deltas);
    decode_boxes<<<(NB * NANCH + 255) / 256, 256, 0, stream>>>(deltas, anchors, boxes);

    // 4) top-k, sort, NMS, compact
    topk_select<<<NB, 1024, 0, stream>>>(scores, boxes, scwork, tk_s, tk_i);
    sort1024<<<NB, 1024, 0, stream>>>(tk_s, tk_i, boxes, sortedb);
    nms_compact<<<NB, 1024, 0, stream>>>(sortedb, rois);

    // 5) ROI align -> pooled [1024][12544] bf16
    roi_align<<<NROI, 256, 0, stream>>>(feat_nhwc, rois, pooled);

    // 6) FC1 / FC2 (WMMA GEMM) + heads
    gemm_bf16_wmma<<<dim3(NROI / 128, NHID / 128), 512, 0, stream>>>(pooled, fc1_wt, fc1_b, h1,
                                                                     NROI, NHID, KFC1, 1);
    gemm_bf16_wmma<<<dim3(NROI / 128, NHID / 128), 512, 0, stream>>>(h1, fc2_wt, fc2_b, h2,
                                                                     NROI, NHID, NHID, 1);
    final_heads<<<NROI / 256, 256, 0, stream>>>(h2, cls_w, cls_b, box_w, box_b, out);
}
